// GCN_Module_8933531975947
// MI455X (gfx1250) — matrix operand order repaired and checked by reading, not compiled, not run on hardware
//
#include <hip/hip_runtime.h>

typedef __attribute__((ext_vector_type(16))) __bf16 v16bf;
typedef __attribute__((ext_vector_type(8)))  __bf16 v8bf;
typedef __attribute__((ext_vector_type(8)))  float  v8f;

union Frag16 { v16bf v; v8bf h[2]; };

#define NTOK 4096   // H*W
#define CCH  256    // channels
#define XW_OFF 4194304ull   // B*C*H*W floats, start of x_w in d_out

__device__ __forceinline__ __bf16 f2bf(float f) {
    union { float f; unsigned u; } a; a.f = f;
    unsigned u = a.u;
    unsigned r = u + 0x7FFFu + ((u >> 16) & 1u);   // round-to-nearest-even
    union { unsigned short s; __bf16 b; } o; o.s = (unsigned short)(r >> 16);
    return o.b;
}

// ---- convert the four 256x256 weights to bf16 (row-major [o][c]) ----
__global__ void wcvt_kernel(const float* __restrict__ wq, const float* __restrict__ wk,
                            const float* __restrict__ wv, const float* __restrict__ wo,
                            __bf16* __restrict__ wbf) {
    int i = blockIdx.x * blockDim.x + threadIdx.x;   // 0..65535
    if (i < 65536) {
        wbf[i]          = f2bf(wq[i]);
        wbf[i +  65536] = f2bf(wk[i]);
        wbf[i + 131072] = f2bf(wv[i]);
        wbf[i + 196608] = f2bf(wo[i]);
    }
}

// ---- x [B][C][N] f32  ->  xt [B][N][C] bf16 (LDS tile transpose) ----
__global__ void xt_kernel(const float* __restrict__ x, __bf16* __restrict__ xt) {
    __shared__ float tile[32][33];
    int b  = blockIdx.z;
    int c0 = blockIdx.y * 32;
    int n0 = blockIdx.x * 32;
    int tx = threadIdx.x, ty = threadIdx.y;          // 32 x 8
    const float* xb = x + (size_t)b * CCH * NTOK;
    #pragma unroll
    for (int k = 0; k < 4; ++k)
        tile[ty + 8*k][tx] = xb[(size_t)(c0 + ty + 8*k) * NTOK + n0 + tx];
    __syncthreads();
    __bf16* xtb = xt + (size_t)b * NTOK * CCH;
    #pragma unroll
    for (int k = 0; k < 4; ++k)
        xtb[(size_t)(n0 + ty + 8*k) * CCH + c0 + tx] = f2bf(tile[tx][ty + 8*k]);
}

// ---- QKV projections via WMMA: Q,K -> [B][N][C] bf16 ; V -> [B][C][N] bf16 ----
__global__ void __launch_bounds__(32) qkv_kernel(const __bf16* __restrict__ xt,
                                                 const __bf16* __restrict__ wbf,
                                                 __bf16* __restrict__ qb,
                                                 __bf16* __restrict__ kb,
                                                 __bf16* __restrict__ vb) {
    const int which = blockIdx.z;        // 0=Q 1=K 2=V
    const int b  = blockIdx.y;
    const int n0 = (blockIdx.x & 255) * 16;
    const int o0 = (blockIdx.x >> 8)  * 16;
    const int lane = threadIdx.x & 31;
    const int l15  = lane & 15;
    const int hi   = lane >> 4;
    const int koffA = hi * 8;            // 16-bit A-frag K offset (ISA 7.12.2)
    const int koffB = hi * 8;            // 16-bit B-frag K offset
    const __bf16* xtb = xt + (size_t)b * NTOK * CCH;
    const __bf16* w   = wbf + (size_t)which * 65536;
    v8f acc = {};
    const __bf16* arow = (which < 2) ? xtb + (size_t)(n0 + l15) * CCH
                                     : w   + (size_t)(o0 + l15) * CCH;
    const __bf16* brow = (which < 2) ? w   + (size_t)(o0 + l15) * CCH
                                     : xtb + (size_t)(n0 + l15) * CCH;
    #pragma unroll
    for (int f = 0; f < 8; ++f) {        // K = 256 = 8 x 32
        Frag16 a, bb;
        a.h[0]  = *(const v8bf*)(arow + 32*f + koffA);
        a.h[1]  = *(const v8bf*)(arow + 32*f + 16 + koffA);
        bb.h[0] = *(const v8bf*)(brow + 32*f + koffB);
        bb.h[1] = *(const v8bf*)(brow + 32*f + koffB + 16);
        acc = __builtin_amdgcn_wmma_f32_16x16x32_bf16(false, a.v, false, bb.v,
                                                      (short)0, acc, false, false);
    }
    if (which < 2) {
        __bf16* dst = (which == 0 ? qb : kb) + (size_t)b * NTOK * CCH;
        #pragma unroll
        for (int r = 0; r < 8; ++r)
            dst[(size_t)(n0 + r + hi*8) * CCH + o0 + l15] = f2bf(acc[r]);
    } else {
        __bf16* dst = vb + (size_t)b * CCH * NTOK;   // channel-major
        #pragma unroll
        for (int r = 0; r < 8; ++r)
            dst[(size_t)(o0 + r + hi*8) * NTOK + n0 + l15] = f2bf(acc[r]);
    }
}

// ---- fused attention: S=sigmoid(QK^T/16) -> x_w, rel=SV/N, Wo, residual, LN ----
__global__ void __launch_bounds__(128) attn_kernel(
        const float* __restrict__ x,
        const __bf16* __restrict__ qb, const __bf16* __restrict__ kb,
        const __bf16* __restrict__ vb, const __bf16* __restrict__ wbf,
        const float* __restrict__ gamma, const float* __restrict__ beta,
        float* __restrict__ out) {
    __shared__ __bf16 Sbuf[2][16][64];   // ping-pong sigmoid(S) block, bf16
    __shared__ __bf16 Rel[16][256];      // rel tile, bf16, feeds Wo GEMM
    __shared__ float  Yf[16][256];       // residual+proj, pre-LN
    __shared__ float  Mu[16], Rs[16];

    const int b    = blockIdx.x >> 8;
    const int n0   = (blockIdx.x & 255) * 16;
    const int tid  = threadIdx.x;
    const int w    = tid >> 5;           // wave 0..3
    const int lane = tid & 31;
    const int l15  = lane & 15;
    const int hi   = lane >> 4;
    const int koffA = hi * 8;
    const int koffB = hi * 8;

    // preload this tile's Q A-fragments (K = 256 -> 8 frags, kept in VGPRs)
    const __bf16* qrow = qb + (size_t)b * NTOK * CCH + (size_t)(n0 + l15) * CCH;
    Frag16 aq[8];
    #pragma unroll
    for (int f = 0; f < 8; ++f) {
        aq[f].h[0] = *(const v8bf*)(qrow + 32*f + koffA);
        aq[f].h[1] = *(const v8bf*)(qrow + 32*f + 16 + koffA);
    }

    const __bf16* kbB = kb + (size_t)b * NTOK * CCH;
    const __bf16* vbB = vb + (size_t)b * CCH * NTOK;
    float* xw = out + XW_OFF + ((size_t)b * NTOK + n0) * NTOK;

    v8f rel0 = {}, rel1 = {}, rel2 = {}, rel3 = {};

    for (int mo = 0; mo < 64; ++mo) {                  // 64-wide m blocks
        const int buf = mo & 1;
        const int m1  = mo * 64 + w * 16;              // this wave's S sub-tile
        const __bf16* krow = kbB + (size_t)(m1 + l15) * CCH;
        v8f s = {};
        #pragma unroll
        for (int f = 0; f < 8; ++f) {
            Frag16 bbk;
            bbk.h[0] = *(const v8bf*)(krow + 32*f + koffB);
            bbk.h[1] = *(const v8bf*)(krow + 32*f + koffB + 16);
            s = __builtin_amdgcn_wmma_f32_16x16x32_bf16(false, aq[f].v, false, bbk.v,
                                                        (short)0, s, false, false);
        }
        if (mo + 1 < 64)                               // gfx1250 global_prefetch_b8
            __builtin_prefetch(kbB + (size_t)(m1 + 64 + l15) * CCH, 0, 1);
        // sigmoid(scale*s): NT-stream f32 tile to x_w (write-once, keep L2 for K/V)
        #pragma unroll
        for (int r = 0; r < 8; ++r) {
            const int row = r + hi * 8;
            float sv = 1.0f / (1.0f + __expf(-s[r] * 0.0625f));
            __builtin_nontemporal_store(sv, &xw[(size_t)row * NTOK + m1 + l15]);
            Sbuf[buf][row][w * 16 + l15] = f2bf(sv);
        }
        __syncthreads();                               // writes(buf) visible; reads(buf^1) done
        // rel += S(16x64) @ V(64 x this wave's 64 channels)
        const int ch0 = w * 64;
        #pragma unroll
        for (int ks = 0; ks < 2; ++ks) {
            Frag16 as;
            const __bf16* srow = &Sbuf[buf][l15][ks * 32];
            as.h[0] = *(const v8bf*)(srow + koffA);
            as.h[1] = *(const v8bf*)(srow + 16 + koffA);
            const size_t mbase = (size_t)mo * 64 + ks * 32 + koffB;
            #pragma unroll
            for (int ct = 0; ct < 4; ++ct) {
                Frag16 bv;
                const __bf16* vrow = vbB + (size_t)(ch0 + ct*16 + l15) * NTOK + mbase;
                bv.h[0] = *(const v8bf*)(vrow);
                bv.h[1] = *(const v8bf*)(vrow + 16);
                v8f& acc = (ct==0) ? rel0 : (ct==1) ? rel1 : (ct==2) ? rel2 : rel3;
                acc = __builtin_amdgcn_wmma_f32_16x16x32_bf16(false, as.v, false, bv.v,
                                                              (short)0, acc, false, false);
            }
        }
        // no trailing barrier: next iteration writes the other Sbuf half
    }
    __syncthreads();

    // rel /= N, deposit bf16 to LDS for the Wo GEMM
    {
        const int ch0 = w * 64;
        #pragma unroll
        for (int ct = 0; ct < 4; ++ct) {
            v8f acc = (ct==0) ? rel0 : (ct==1) ? rel1 : (ct==2) ? rel2 : rel3;
            #pragma unroll
            for (int r = 0; r < 8; ++r)
                Rel[r + hi*8][ch0 + ct*16 + l15] = f2bf(acc[r] * (1.0f / 4096.0f));
        }
    }
    __syncthreads();

    // Wo projection (WMMA) + residual from original f32 x
    const __bf16* wo = wbf + 196608;
    const float*  xb = x + (size_t)b * CCH * NTOK;
    #pragma unroll
    for (int ct = 0; ct < 4; ++ct) {
        const int oc0 = w * 64 + ct * 16;
        const __bf16* brow = wo + (size_t)(oc0 + l15) * CCH;
        v8f acc = {};
        #pragma unroll
        for (int f = 0; f < 8; ++f) {
            Frag16 ar, bb;
            const __bf16* rrow = &Rel[l15][32 * f];
            ar.h[0] = *(const v8bf*)(rrow + koffA);
            ar.h[1] = *(const v8bf*)(rrow + 16 + koffA);
            bb.h[0] = *(const v8bf*)(brow + 32*f + koffB);
            bb.h[1] = *(const v8bf*)(brow + 32*f + koffB + 16);
            acc = __builtin_amdgcn_wmma_f32_16x16x32_bf16(false, ar.v, false, bb.v,
                                                          (short)0, acc, false, false);
        }
        #pragma unroll
        for (int r = 0; r < 8; ++r) {
            const int row = r + hi * 8;
            const int c   = oc0 + l15;
            Yf[row][c] = acc[r] + xb[(size_t)c * NTOK + n0 + row];
        }
    }
    __syncthreads();

    // LayerNorm over channels for each of the 16 rows
    if (tid < 16) {
        float sum = 0.f;
        for (int c = 0; c < 256; ++c) sum += Yf[tid][c];
        float mu = sum * (1.0f / 256.0f);
        float var = 0.f;
        for (int c = 0; c < 256; ++c) { float d = Yf[tid][c] - mu; var += d * d; }
        Mu[tid] = mu;
        Rs[tid] = rsqrtf(var * (1.0f / 256.0f) + 1e-5f);
    }
    __syncthreads();

    float* ob = out + (size_t)b * CCH * NTOK;
    for (int c = tid; c < 256; c += 128) {
        const float g = gamma[c], be = beta[c];
        #pragma unroll
        for (int r = 0; r < 16; ++r) {
            float yv = (Yf[r][c] - Mu[r]) * Rs[r] * g + be;
            __builtin_nontemporal_store(yv, &ob[(size_t)c * NTOK + n0 + r]);
        }
    }
}

extern "C" void kernel_launch(void* const* d_in, const int* in_sizes, int n_in,
                              void* d_out, int out_size, void* d_ws, size_t ws_size,
                              hipStream_t stream) {
    const float* x     = (const float*)d_in[0];
    const float* Wq    = (const float*)d_in[1];
    const float* Wk    = (const float*)d_in[2];
    const float* Wv    = (const float*)d_in[3];
    const float* Wo    = (const float*)d_in[4];
    const float* gamma = (const float*)d_in[5];
    const float* beta  = (const float*)d_in[6];
    float* out = (float*)d_out;

    char* ws = (char*)d_ws;
    __bf16* xt  = (__bf16*)(ws);                 //  8 MB  [B][N][C]
    __bf16* qb  = (__bf16*)(ws + ( 8u << 20));   //  8 MB  [B][N][C]
    __bf16* kb  = (__bf16*)(ws + (16u << 20));   //  8 MB  [B][N][C]
    __bf16* vb  = (__bf16*)(ws + (24u << 20));   //  8 MB  [B][C][N]
    __bf16* wbf = (__bf16*)(ws + (32u << 20));   // 512 KB [Wq|Wk|Wv|Wo]

    wcvt_kernel<<<256, 256, 0, stream>>>(Wq, Wk, Wv, Wo, wbf);
    xt_kernel<<<dim3(128, 8, 4), dim3(32, 8), 0, stream>>>(x, xt);
    qkv_kernel<<<dim3(4096, 4, 3), 32, 0, stream>>>(xt, wbf, qb, kb, vb);
    attn_kernel<<<dim3(1024), 128, 0, stream>>>(x, qb, kb, vb, wbf, gamma, beta, out);
}
